// GMM_44306882626243
// MI455X (gfx1250) — matrix operand order repaired and checked by reading, not compiled, not run on hardware
//
#include <hip/hip_runtime.h>

typedef __attribute__((ext_vector_type(16))) __bf16 bf16x16;
typedef __attribute__((ext_vector_type(8)))  float  floatx8;

#define D 256
#define KC 64
#define NPTS 8192

// ---------------------------------------------------------------------------
// Kernel 1: per-cluster in-place Gauss-Jordan inversion of var_k in LDS
// (SPD, well-conditioned -> no pivoting). Produces bf16 hi/lo split of the
// precision matrix plus constk[k] = log(pi_k) + 0.5*logdet(P_k) - 0.5*D*log(2pi).
// One workgroup (8 waves) per cluster; 256KB matrix + 1KB column buffer in LDS.
// ---------------------------------------------------------------------------
__global__ void __launch_bounds__(256) gmm_invert(
    const float* __restrict__ var, const float* __restrict__ pi,
    __bf16* __restrict__ Phi, __bf16* __restrict__ Plo,
    float* __restrict__ constk)
{
    __shared__ float M[D * D];
    __shared__ float colbuf[D];
    const int t = threadIdx.x;          // thread t owns column t
    const int k = blockIdx.x;

    const float* src = var + ((size_t)k << 16);
    for (int i = t; i < D * D; i += 256) M[i] = src[i];
    __syncthreads();

    float logdet = 0.0f;                 // log det(var_k), thread 0 only
    for (int p = 0; p < D; ++p) {
        const float piv  = M[p * D + p];
        const float ipiv = 1.0f / piv;
        if (t == 0) logdet += logf(piv);
        colbuf[t]          = M[t * D + p];   // stash pivot column
        const float rowp_t = M[p * D + t];   // pivot row element (my column)
        __syncthreads();                      // colbuf visible, reads done

        const float nrow = (t == p) ? ipiv : rowp_t * ipiv;
        if (t == p) {
            #pragma unroll 4
            for (int r = 0; r < D; ++r)
                M[r * D + t] = (r == p) ? ipiv : (-colbuf[r] * ipiv);
        } else {
            #pragma unroll 4
            for (int r = 0; r < D; ++r)
                M[r * D + t] = (r == p) ? nrow
                                        : fmaf(-colbuf[r], nrow, M[r * D + t]);
        }
        __syncthreads();                      // updates visible for next pivot
    }

    // write inverse as bf16 hi/lo split planes
    __bf16* oh = Phi + ((size_t)k << 16);
    __bf16* ol = Plo + ((size_t)k << 16);
    for (int i = t; i < D * D; i += 256) {
        const float v = M[i];
        const __bf16 h = (__bf16)v;
        oh[i] = h;
        ol[i] = (__bf16)(v - (float)h);
    }
    if (t == 0) {
        const float LOG2PI = 1.8378770664093453f;
        // log det(P) = -log det(var)
        constk[k] = logf(pi[k]) - 0.5f * logdet - 0.5f * (float)D * LOG2PI;
    }
}

// ---------------------------------------------------------------------------
// Kernel 2: fused scores.  One wave = 16 rows of x against one cluster k.
// quad = rowsum( (Xm @ P_k) .* Xm ), accumulated over 16x16 output tiles with
// V_WMMA_F32_16X16X32_BF16 in compensated bf16 (hi/lo split, 3 WMMAs/chunk).
// B fragments come from rows of P (symmetry) -> contiguous 32B bf16 loads
// giving k = 16*half + s per lane, the same k order as the A fragments.
// ---------------------------------------------------------------------------
__global__ void __launch_bounds__(256) gmm_score(
    const float* __restrict__ x, const float* __restrict__ mu,
    const __bf16* __restrict__ Phi, const __bf16* __restrict__ Plo,
    const float* __restrict__ constk, float* __restrict__ scores)
{
    const int lane   = threadIdx.x & 31;
    const int wave   = threadIdx.x >> 5;
    const int k      = blockIdx.y;
    const int ntbase = (blockIdx.x * 8 + wave) * 16;
    const int m      = lane & 15;        // A row / B column / C column index
    const int half   = lane >> 4;        // K-group select for A and B
    const int g      = half * 16;        // A-layout K offset within chunk

    // ---- build resident A fragments (xm tile, 16x256) split into bf16 hi/lo
    bf16x16 a_hi[8], a_lo[8];
    const float* xrow  = x  + (size_t)(ntbase + m) * D;
    const float* murow = mu + (size_t)k * D;
    #pragma unroll
    for (int c = 0; c < 8; ++c) {        // 8 chunks of K=32 along D
        const int d0 = c * 32 + g;
        #pragma unroll
        for (int h2 = 0; h2 < 2; ++h2) {             // runs K=[g,g+7], [g+8..g+15]
            const int db = d0 + h2 * 8;
            #pragma unroll
            for (int q4 = 0; q4 < 2; ++q4) {
                const float4 xv = *(const float4*)(xrow  + db + q4 * 4);
                const float4 mv = *(const float4*)(murow + db + q4 * 4);
                const float fv[4] = { xv.x - mv.x, xv.y - mv.y,
                                      xv.z - mv.z, xv.w - mv.w };
                #pragma unroll
                for (int e = 0; e < 4; ++e) {
                    const int idx = h2 * 8 + q4 * 4 + e;
                    const float v = fv[e];
                    const __bf16 hh = (__bf16)v;
                    a_hi[c][idx] = hh;
                    a_lo[c][idx] = (__bf16)(v - (float)hh);
                }
            }
        }
    }

    float quadp[8] = {0.f, 0.f, 0.f, 0.f, 0.f, 0.f, 0.f, 0.f};
    const __bf16* phiBase = Phi + ((size_t)k << 16);
    const __bf16* ploBase = Plo + ((size_t)k << 16);

    for (int j = 0; j < 16; ++j) {       // output column tiles (16 wide)
        floatx8 acc = {};
        const int brow = j * 16 + m;     // P symmetric: column block = row block
        const __bf16* bh_ptr = phiBase + (size_t)brow * D + half * 16;
        const __bf16* bl_ptr = ploBase + (size_t)brow * D + half * 16;
        #pragma unroll
        for (int c = 0; c < 8; ++c) {
            const bf16x16 bh = *(const bf16x16*)(bh_ptr + c * 32);
            const bf16x16 bl = *(const bf16x16*)(bl_ptr + c * 32);
            acc = __builtin_amdgcn_wmma_f32_16x16x32_bf16(
                      false, a_hi[c], false, bh, (short)0, acc, false, false);
            acc = __builtin_amdgcn_wmma_f32_16x16x32_bf16(
                      false, a_hi[c], false, bl, (short)0, acc, false, false);
            acc = __builtin_amdgcn_wmma_f32_16x16x32_bf16(
                      false, a_lo[c], false, bh, (short)0, acc, false, false);
        }
        // fuse:  quad[M] += Y[M][n] * xm[M][j*16+n]   (C layout: M = r+8*half)
        const int   d   = j * 16 + m;
        const float muv = murow[d];
        #pragma unroll
        for (int r = 0; r < 8; ++r) {
            const float xv = x[(size_t)(ntbase + r + 8 * half) * D + d];
            quadp[r] += acc[r] * (xv - muv);
        }
    }

    // reduce over the 16 columns (lanes within each half-wave), emit scores
    const float ck = constk[k];
    #pragma unroll
    for (int r = 0; r < 8; ++r) {
        float q = quadp[r];
        q += __shfl_xor(q, 1, 32);
        q += __shfl_xor(q, 2, 32);
        q += __shfl_xor(q, 4, 32);
        q += __shfl_xor(q, 8, 32);
        if (m == r) {
            const int row = ntbase + r + 8 * half;
            scores[(size_t)row * KC + k] = ck - 0.5f * q;
        }
    }
}

// ---------------------------------------------------------------------------
// Kernel 3: one wave per row, argmax over K=64 scores (first-max tiebreak,
// matching jnp.argmax). Output int32 indices.
// ---------------------------------------------------------------------------
__global__ void __launch_bounds__(256) gmm_argmax(
    const float* __restrict__ scores, int* __restrict__ out)
{
    const int lane = threadIdx.x & 31;
    const int wave = threadIdx.x >> 5;
    const int row  = blockIdx.x * 8 + wave;
    const float* s = scores + (size_t)row * KC;
    const float b0 = s[lane];
    const float b1 = s[lane + 32];
    float best = b0; int bi = lane;
    if (b1 > best) { best = b1; bi = lane + 32; }
    #pragma unroll
    for (int off = 16; off > 0; off >>= 1) {
        const float ob = __shfl_xor(best, off, 32);
        const int   oi = __shfl_xor(bi,   off, 32);
        if (ob > best || (ob == best && oi < bi)) { best = ob; bi = oi; }
    }
    if (lane == 0) out[row] = bi;
}

// ---------------------------------------------------------------------------
extern "C" void kernel_launch(void* const* d_in, const int* in_sizes, int n_in,
                              void* d_out, int out_size, void* d_ws, size_t ws_size,
                              hipStream_t stream) {
    const float* x   = (const float*)d_in[0];   // (8192, 256)
    const float* mu  = (const float*)d_in[1];   // (64, 256)
    const float* var = (const float*)d_in[2];   // (64, 256, 256)
    const float* pi  = (const float*)d_in[3];   // (64,)

    char* ws = (char*)d_ws;
    __bf16* Phi    = (__bf16*)ws;                                // 8 MB
    __bf16* Plo    = (__bf16*)(ws + (size_t)(8u << 20));         // 8 MB
    float*  constk = (float*)(ws + (size_t)(16u << 20));         // 256 B
    float*  scores = (float*)(ws + (size_t)(16u << 20) + 1024);  // 2 MB
    int*    out    = (int*)d_out;

    gmm_invert<<<dim3(KC), dim3(256), 0, stream>>>(var, pi, Phi, Plo, constk);
    gmm_score<<<dim3(NPTS / 128, KC), dim3(256), 0, stream>>>(
        x, mu, Phi, Plo, constk, scores);
    gmm_argmax<<<dim3(NPTS / 8), dim3(256), 0, stream>>>(scores, out);
}
